// CrossAttention_35905926595226
// MI455X (gfx1250) — compile-verified, matrix operand order checked by reading, not run on hardware
//
#include <hip/hip_runtime.h>
#include <hip/hip_bf16.h>

#define DMODEL 768
#define LDP (DMODEL + 8)        // padded LDS leading dim (halves)
#define DK 64
#define NHEADS 12
#define SEQ 64
#define LN_EPS 1e-5f

typedef __attribute__((ext_vector_type(16))) _Float16 v16h;
typedef __attribute__((ext_vector_type(8)))  _Float16 v8h;
typedef __attribute__((ext_vector_type(8)))  float    v8f;

// ---------------------------------------------------------------------------
// WMMA helpers (CDNA5 wave32, V_WMMA_F32_16X16X32_F16)
// ---------------------------------------------------------------------------
__device__ __forceinline__ v8f wmma_f16(v16h a, v16h b, v8f c) {
  return __builtin_amdgcn_wmma_f32_16x16x32_f16(false, a, false, b, (short)0, c,
                                                false, false);
}

// 16x32 f16 operand tile in the ISA A-operand layout from a row-major f16
// matrix (leading dim ld).  Also used for B operand via the transposed matrix.
__device__ __forceinline__ v16h load_tile16x32(const _Float16* __restrict__ base,
                                               int ld, int major0, int k0) {
  int lane = threadIdx.x & 31;
  int r  = major0 + (lane & 15);
  int kb = k0 + ((lane >> 4) << 3);
  const _Float16* p = base + (size_t)r * ld + kb;
  v8h lo = *(const v8h*)(p);
  v8h hi = *(const v8h*)(p + 16);
  v16h a;
#pragma unroll
  for (int t = 0; t < 8; ++t) { a[t] = lo[t]; a[t + 8] = hi[t]; }
  return a;
}

// ---------------------------------------------------------------------------
// Kernel 1: weights fp32 [in][out] -> f16 [out][in] (transposed)
// ---------------------------------------------------------------------------
__global__ __launch_bounds__(256) void wconv_kernel(
    const float* __restrict__ W0, const float* __restrict__ W1,
    const float* __restrict__ W2, const float* __restrict__ W3,
    _Float16* __restrict__ T0, _Float16* __restrict__ T1,
    _Float16* __restrict__ T2, _Float16* __restrict__ T3) {
  int idx   = blockIdx.x * 256 + threadIdx.x;
  int which = idx / (DMODEL * DMODEL);
  int rem   = idx % (DMODEL * DMODEL);
  int k = rem / DMODEL, n = rem % DMODEL;
  const float* W = (which == 0) ? W0 : (which == 1) ? W1 : (which == 2) ? W2 : W3;
  _Float16*    T = (which == 0) ? T0 : (which == 1) ? T1 : (which == 2) ? T2 : T3;
  T[(size_t)n * DMODEL + k] = (_Float16)W[(size_t)k * DMODEL + n];
}

// ---------------------------------------------------------------------------
// Shared GEMM inner: C[2 m-tiles] over K=768, A panel in LDS (32 x LDP f16),
// B from global transposed weights.  k unrolled x2 so two B loads in flight.
// ---------------------------------------------------------------------------
__device__ __forceinline__ void gemm32_panel(const _Float16* As,  // LDS panel
                                             const _Float16* __restrict__ WT,
                                             int col0, v8f& c0, v8f& c1) {
  for (int k0 = 0; k0 < DMODEL; k0 += 64) {
    v16h b0  = load_tile16x32(WT, DMODEL, col0, k0);
    v16h b1  = load_tile16x32(WT, DMODEL, col0, k0 + 32);
    v16h a00 = load_tile16x32(As, LDP, 0,  k0);
    v16h a10 = load_tile16x32(As, LDP, 16, k0);
    v16h a01 = load_tile16x32(As, LDP, 0,  k0 + 32);
    v16h a11 = load_tile16x32(As, LDP, 16, k0 + 32);
    c0 = wmma_f16(a00, b0, c0);
    c1 = wmma_f16(a10, b0, c1);
    c0 = wmma_f16(a01, b1, c0);
    c1 = wmma_f16(a11, b1, c1);
  }
}

// ---------------------------------------------------------------------------
// Kernel 2: K/V projections of h.  Block = 32 rows of h staged f16 in LDS.
// 8 waves sweep 96 N-tiles (48 K row-major + 48 V stored transposed as
// Vt[bc][head][d][j]).
// ---------------------------------------------------------------------------
__global__ __launch_bounds__(256) void kv_proj_kernel(
    const float* __restrict__ h,        // [bc*64, 768]
    const _Float16* __restrict__ WkT, const float* __restrict__ bk,
    const _Float16* __restrict__ WvT, const float* __restrict__ bv,
    _Float16* __restrict__ K16,         // [bc*64, 768]
    _Float16* __restrict__ Vt16) {      // [bc][12][64(d)][64(j)]
  __shared__ _Float16 hs[32][LDP];
  int row0 = blockIdx.x * 32;
  int bc   = row0 / SEQ;
  int j0   = row0 % SEQ;                // 0 or 32
  for (int t = threadIdx.x; t < 32 * DMODEL; t += 256) {
    int r = t / DMODEL, cidx = t % DMODEL;
    hs[r][cidx] = (_Float16)h[(size_t)(row0 + r) * DMODEL + cidx];
  }
  __syncthreads();
  int wave = threadIdx.x >> 5, lane = threadIdx.x & 31;
  for (int tt = wave; tt < 96; tt += 8) {
    int isV  = (tt >= 48);
    int col0 = (isV ? tt - 48 : tt) * 16;
    v8f c0 = {}, c1 = {};
    gemm32_panel(&hs[0][0], isV ? WvT : WkT, col0, c0, c1);
    int ccol  = col0 + (lane & 15);
    int rbase = (lane >> 4) << 3;
    float bias = isV ? bv[ccol] : bk[ccol];
    if (!isV) {
#pragma unroll
      for (int r = 0; r < 8; ++r) {
        K16[(size_t)(row0 + rbase + r) * DMODEL + ccol]      = (_Float16)(c0[r] + bias);
        K16[(size_t)(row0 + 16 + rbase + r) * DMODEL + ccol] = (_Float16)(c1[r] + bias);
      }
    } else {
      int head = ccol / DK, d = ccol % DK;
      _Float16* dst =
          Vt16 + (((size_t)bc * NHEADS + head) * DK + d) * SEQ + j0 + rbase;
      v8h o0, o1;
#pragma unroll
      for (int r = 0; r < 8; ++r) {
        o0[r] = (_Float16)(c0[r] + bias);
        o1[r] = (_Float16)(c1[r] + bias);
      }
      *(v8h*)dst        = o0;   // j = j0+rbase .. +7
      *(v8h*)(dst + 16) = o1;   // j = j0+16+rbase .. +7
    }
  }
}

// ---------------------------------------------------------------------------
// Kernel 3: fused LayerNorm(e) + Q projection.  Block = 32 rows.
// Softmax scale 1/sqrt(DK)=0.125 folded into the Q store.
// ---------------------------------------------------------------------------
__global__ __launch_bounds__(256) void lnq_kernel(
    const float* __restrict__ e, const float* __restrict__ ln_g,
    const float* __restrict__ ln_b, const _Float16* __restrict__ WqT,
    const float* __restrict__ bq, _Float16* __restrict__ Q16) {
  __shared__ _Float16 es[32][LDP];
  __shared__ float red[32][9], red2[32][9];
  __shared__ float mu_s[32], rs_s[32];
  int row0 = blockIdx.x * 32;
  int rloc = threadIdx.x >> 3;   // 0..31
  int tin  = threadIdx.x & 7;    // 8 threads per row, 96 elems each
  {
    const float* rowp = e + (size_t)(row0 + rloc) * DMODEL + tin * 96;
    float s = 0.f, s2 = 0.f;
    for (int cidx = 0; cidx < 96; ++cidx) {
      float v = rowp[cidx];
      s += v; s2 += v * v;
    }
    red[rloc][tin] = s; red2[rloc][tin] = s2;
  }
  __syncthreads();
  if (tin == 0) {
    float a = 0.f, b2 = 0.f;
    for (int t = 0; t < 8; ++t) { a += red[rloc][t]; b2 += red2[rloc][t]; }
    float mu  = a / DMODEL;
    float var = b2 / DMODEL - mu * mu;
    mu_s[rloc] = mu;
    rs_s[rloc] = rsqrtf(var + LN_EPS);
  }
  __syncthreads();
  for (int t = threadIdx.x; t < 32 * DMODEL; t += 256) {
    int r = t / DMODEL, cidx = t % DMODEL;
    float v = e[(size_t)(row0 + r) * DMODEL + cidx];
    es[r][cidx] =
        (_Float16)(((v - mu_s[r]) * rs_s[r]) * ln_g[cidx] + ln_b[cidx]);
  }
  __syncthreads();
  int wave = threadIdx.x >> 5, lane = threadIdx.x & 31;
  for (int nt = wave; nt < 48; nt += 8) {
    int col0 = nt * 16;
    v8f c0 = {}, c1 = {};
    gemm32_panel(&es[0][0], WqT, col0, c0, c1);
    int ccol  = col0 + (lane & 15);
    int rbase = (lane >> 4) << 3;
    float bias = bq[ccol];
#pragma unroll
    for (int r = 0; r < 8; ++r) {
      Q16[(size_t)(row0 + rbase + r) * DMODEL + ccol] =
          (_Float16)((c0[r] + bias) * 0.125f);
      Q16[(size_t)(row0 + 16 + rbase + r) * DMODEL + ccol] =
          (_Float16)((c1[r] + bias) * 0.125f);
    }
  }
}

// ---------------------------------------------------------------------------
// Kernel 4: attention per (b,c,n) x head.  Block = 128 threads (4 waves),
// wave w owns query rows [16w,16w+16).
// ---------------------------------------------------------------------------
__global__ __launch_bounds__(128) void attn_kernel(
    const _Float16* __restrict__ Q16,   // [bcn*64, 768] pre-scaled
    const _Float16* __restrict__ K16,   // [bc*64, 768]
    const _Float16* __restrict__ Vt16,  // [bc][12][64][64]
    _Float16* __restrict__ O16) {       // [bcn*64, 768]
  __shared__ float    Ss[4][16][SEQ];
  __shared__ _Float16 Ps[4][16][SEQ + 8];
  int bcn = blockIdx.x, head = blockIdx.y;
  int bc  = bcn >> 2;
  int wave = threadIdx.x >> 5, lane = threadIdx.x & 31;
  int qrow0 = bcn * SEQ + wave * 16;
  int krow0 = bc * SEQ;
  int ccolb = (lane & 15);
  int rbase = (lane >> 4) << 3;
  // S = Q @ K^T : A (Q) hoisted, 4 j-tiles of B (K row-major = Kt B-operand)
  v16h qa0 = load_tile16x32(Q16, DMODEL, qrow0, head * DK);
  v16h qa1 = load_tile16x32(Q16, DMODEL, qrow0, head * DK + 32);
#pragma unroll
  for (int jt = 0; jt < 4; ++jt) {
    v16h b0 = load_tile16x32(K16, DMODEL, krow0 + jt * 16, head * DK);
    v16h b1 = load_tile16x32(K16, DMODEL, krow0 + jt * 16, head * DK + 32);
    v8f c = {};
    c = wmma_f16(qa0, b0, c);
    c = wmma_f16(qa1, b1, c);
#pragma unroll
    for (int r = 0; r < 8; ++r) Ss[wave][rbase + r][jt * 16 + ccolb] = c[r];
  }
  __syncthreads();
  // softmax over j (one lane per row; two passes to stay register-light)
  if (lane < 16) {
    float m = -1e30f;
    for (int j = 0; j < SEQ; ++j) m = fmaxf(m, Ss[wave][lane][j]);
    float sum = 0.f;
    for (int j = 0; j < SEQ; ++j) sum += __expf(Ss[wave][lane][j] - m);
    float inv = 1.f / sum;
    for (int j = 0; j < SEQ; ++j)
      Ps[wave][lane][j] = (_Float16)(__expf(Ss[wave][lane][j] - m) * inv);
  }
  __syncthreads();
  // O = P @ V : A (P) hoisted from LDS, B (Vt) contiguous
  v16h pa0 = load_tile16x32(&Ps[wave][0][0], SEQ + 8, 0, 0);
  v16h pa1 = load_tile16x32(&Ps[wave][0][0], SEQ + 8, 0, 32);
  const _Float16* vbase = Vt16 + (((size_t)bc * NHEADS + head) * DK) * SEQ;
#pragma unroll
  for (int dt = 0; dt < 4; ++dt) {
    v16h b0 = load_tile16x32(vbase, SEQ, dt * 16, 0);
    v16h b1 = load_tile16x32(vbase, SEQ, dt * 16, 32);
    v8f c = {};
    c = wmma_f16(pa0, b0, c);
    c = wmma_f16(pa1, b1, c);
    int ccol = head * DK + dt * 16 + ccolb;
#pragma unroll
    for (int r = 0; r < 8; ++r)
      O16[(size_t)(qrow0 + rbase + r) * DMODEL + ccol] = (_Float16)c[r];
  }
}

// ---------------------------------------------------------------------------
// Kernel 5: out = O @ Wo + bo + e  (fp32 result).  The O panel is already
// f16, so stage it into LDS with the gfx1250 async VMEM->LDS path
// (GLOBAL_LOAD_ASYNC_TO_LDS_B128, tracked by ASYNCcnt).
// ---------------------------------------------------------------------------
__global__ __launch_bounds__(256) void out_proj_kernel(
    const _Float16* __restrict__ O16, const _Float16* __restrict__ WoT,
    const float* __restrict__ bo, const float* __restrict__ e,
    float* __restrict__ out) {
  __shared__ _Float16 os[32][LDP];
  int row0 = blockIdx.x * 32;
  // 32 rows x 768 halves = 3072 16B chunks; each thread issues 12 async copies
  for (int i = threadIdx.x; i < 32 * 96; i += 256) {
    int r = i / 96, cc = (i % 96) * 8;
    unsigned lds_off = (unsigned)(uintptr_t)(&os[r][cc]);
    const _Float16* gp = O16 + (size_t)(row0 + r) * DMODEL + cc;
    asm volatile("global_load_async_to_lds_b128 %0, %1, off"
                 :: "v"(lds_off), "v"(gp)
                 : "memory");
  }
  asm volatile("s_wait_asynccnt 0x0" ::: "memory");
  __syncthreads();
  int wave = threadIdx.x >> 5, lane = threadIdx.x & 31;
  for (int nt = wave; nt < 48; nt += 8) {
    int col0 = nt * 16;
    v8f c0 = {}, c1 = {};
    gemm32_panel(&os[0][0], WoT, col0, c0, c1);
    int ccol  = col0 + (lane & 15);
    int rbase = (lane >> 4) << 3;
    float bias = bo[ccol];
#pragma unroll
    for (int r = 0; r < 8; ++r) {
      size_t i0 = (size_t)(row0 + rbase + r) * DMODEL + ccol;
      size_t i1 = (size_t)(row0 + 16 + rbase + r) * DMODEL + ccol;
      out[i0] = c0[r] + bias + e[i0];
      out[i1] = c1[r] + bias + e[i1];
    }
  }
}

// ---------------------------------------------------------------------------
extern "C" void kernel_launch(void* const* d_in, const int* in_sizes, int n_in,
                              void* d_out, int out_size, void* d_ws,
                              size_t ws_size, hipStream_t stream) {
  const float* e   = (const float*)d_in[0];   // [8,32,4,64,768]
  const float* h   = (const float*)d_in[1];   // [8,32,64,768]
  const float* Wq  = (const float*)d_in[2];
  const float* bq  = (const float*)d_in[3];
  const float* Wk  = (const float*)d_in[4];
  const float* bk  = (const float*)d_in[5];
  const float* Wv  = (const float*)d_in[6];
  const float* bv  = (const float*)d_in[7];
  const float* Wo  = (const float*)d_in[8];
  const float* bo  = (const float*)d_in[9];
  const float* lng = (const float*)d_in[10];
  const float* lnb = (const float*)d_in[11];
  float* out = (float*)d_out;

  const int ROWS_E = in_sizes[0] / DMODEL;    // 65536
  const int ROWS_H = in_sizes[1] / DMODEL;    // 16384
  const int BCN = ROWS_E / SEQ;               // 1024
  (void)n_in; (void)out_size; (void)ws_size;

  char* ws = (char*)d_ws;
  const size_t WW = (size_t)DMODEL * DMODEL;
  _Float16* WqT  = (_Float16*)ws; ws += WW * 2;
  _Float16* WkT  = (_Float16*)ws; ws += WW * 2;
  _Float16* WvT  = (_Float16*)ws; ws += WW * 2;
  _Float16* WoT  = (_Float16*)ws; ws += WW * 2;
  _Float16* K16  = (_Float16*)ws; ws += (size_t)ROWS_H * DMODEL * 2;
  _Float16* Vt16 = (_Float16*)ws; ws += (size_t)ROWS_H * DMODEL * 2;
  _Float16* Q16  = (_Float16*)ws; ws += (size_t)ROWS_E * DMODEL * 2;
  _Float16* O16  = (_Float16*)ws; ws += (size_t)ROWS_E * DMODEL * 2;

  wconv_kernel<<<(4 * WW) / 256, 256, 0, stream>>>(Wq, Wk, Wv, Wo,
                                                   WqT, WkT, WvT, WoT);
  kv_proj_kernel<<<ROWS_H / 32, 256, 0, stream>>>(h, WkT, bk, WvT, bv,
                                                  K16, Vt16);
  lnq_kernel<<<ROWS_E / 32, 256, 0, stream>>>(e, lng, lnb, WqT, bq, Q16);
  attn_kernel<<<dim3(BCN, NHEADS), 128, 0, stream>>>(Q16, K16, Vt16, O16);
  out_proj_kernel<<<ROWS_E / 32, 256, 0, stream>>>(O16, WoT, bo, e, out);
}
